// Attention_20375324852422
// MI455X (gfx1250) — matrix operand order repaired and checked by reading, not compiled, not run on hardware
//
#include <hip/hip_runtime.h>

// ---------------------------------------------------------------------------
// Problem constants (from the reference)
// ---------------------------------------------------------------------------
constexpr int cB = 2, cT = 2048, cD = 2560, cN = 8, cK = 4, cH = 256;
constexpr int cWIN = 1024;
constexpr float cKMASK = -2.3819763e38f;

// wave32 WMMA vector types (probe-confirmed signatures)
typedef __attribute__((ext_vector_type(16))) __bf16          bf16x16;
typedef __attribute__((ext_vector_type(8)))  float           floatx8;
typedef __attribute__((ext_vector_type(8)))  unsigned short  ushort8;
typedef __attribute__((ext_vector_type(4)))  unsigned int    uint32x4;
typedef __attribute__((ext_vector_type(8)))  unsigned int    uint32x8;

union ABf { bf16x16 v; ushort8 h[2]; };

static __device__ __forceinline__ unsigned short f2bf(float f) {
  union { float f; unsigned u; } x{f};
  unsigned r = x.u + 0x7FFFu + ((x.u >> 16) & 1u);   // round-to-nearest-even
  return (unsigned short)(r >> 16);
}

// LDS byte offset of a generic pointer to __shared__ (addr[31:0] is the LDS
// offset per the flat-aperture mapping).
static __device__ __forceinline__ unsigned lds_off(const void* p) {
  return (unsigned)(unsigned long long)p;
}

// ---- CDNA5 async load: global -> LDS, no VGPR data, ASYNCcnt tracked ------
static __device__ __forceinline__ void async_load_b128(unsigned ldsaddr,
                                                       unsigned byte_off,
                                                       const void* sbase) {
  asm volatile("global_load_async_to_lds_b128 %0, %1, %2"
               :: "v"(ldsaddr), "v"(byte_off), "s"(sbase) : "memory");
}
static __device__ __forceinline__ void wait_asynccnt0() {
  asm volatile("s_wait_asynccnt 0" ::: "memory");
}

// ---- CDNA5 Tensor Data Mover: 2-D tile DMA, TENSORcnt tracked -------------
// Descriptor per ISA 8.3/8.4: group0 = {count/lds_addr/global_addr/type},
// group1 = {flags, tensor dims, tile dims, strides}. data_size = 2 bytes.
static __device__ __forceinline__ void tdm_load_tile_2d(
    unsigned lds_byte, const void* gptr,
    unsigned tensor_d0, unsigned tensor_d1, unsigned long long stride0_elems,
    unsigned tile_d0, unsigned tile_d1,
    unsigned pad_interval_code, unsigned pad_amount_code, unsigned pad_en) {
  const unsigned long long ga = (unsigned long long)gptr;
  uint32x4 g0;
  g0[0] = 1u;                                           // count=1, no gather
  g0[1] = lds_byte;                                     // lds_addr
  g0[2] = (unsigned)ga;                                 // global_addr[31:0]
  g0[3] = (unsigned)((ga >> 32) & 0x01FFFFFFu) | (2u << 30);  // [56:32] | type=2
  uint32x8 g1;
  g1[0] = (1u << 16)                                    // data_size = 2 bytes
        | (pad_en << 20)
        | (pad_interval_code << 22)
        | (pad_amount_code << 25);                      // wg_mask=0 (no cluster)
  g1[1] = (tensor_d0 & 0xFFFFu) << 16;                  // tensor_dim0[15:0]
  g1[2] = (tensor_d0 >> 16) | ((tensor_d1 & 0xFFFFu) << 16);
  g1[3] = (tensor_d1 >> 16) | (tile_d0 << 16);          // tile_dim0
  g1[4] = tile_d1;                                      // tile_dim1, tile_dim2=0
  g1[5] = (unsigned)stride0_elems;                      // tensor_dim0_stride lo
  g1[6] = (unsigned)(stride0_elems >> 32) & 0xFFFFu;    // stride hi, stride1=0
  g1[7] = 0u;
  asm volatile("tensor_load_to_lds %0, %1" :: "s"(g0), "s"(g1) : "memory");
}

// ---------------------------------------------------------------------------
// fp32 -> bf16 bulk convert
// ---------------------------------------------------------------------------
__global__ __launch_bounds__(256) void f2bf_kernel(const float* __restrict__ in,
                                                   unsigned short* __restrict__ out,
                                                   long n) {
  for (long i = blockIdx.x * (long)blockDim.x + threadIdx.x; i < n;
       i += (long)gridDim.x * blockDim.x)
    out[i] = f2bf(in[i]);
}

// ---------------------------------------------------------------------------
// Batched bf16 GEMM: C[z] = A[z] (MxKd) * B[z] (Kd x Ncols), fp32 accum.
// Block tile 128x64, 4 waves, each wave -> 32x64 (2x4 WMMA tiles of 16x16).
// A staged via CDNA5 async global->LDS copies (no VGPR round-trip); B staged
// TRANSPOSED so B fragments are contiguous 16-byte ds_load_b128 reads.
// ---------------------------------------------------------------------------
__global__ __launch_bounds__(128) void gemm_bf16_kernel(
    const unsigned short* __restrict__ A, long sAz, int lda,
    const unsigned short* __restrict__ Bw, long sBz, int ldb,
    float* __restrict__ C, long sCz, int ldc,
    int Kdim) {
  constexpr int BM = 128, BK = 32;
  __shared__ unsigned short As[BM][BK + 8];   // +8 pad vs bank conflicts
  __shared__ unsigned short BsT[64][BK + 8];  // [col][k]

  A  += (long)blockIdx.z * sAz;
  Bw += (long)blockIdx.z * sBz;
  C  += (long)blockIdx.z * sCz;

  const int m0 = blockIdx.x * BM;
  const int n0 = blockIdx.y * 64;
  const int tid  = threadIdx.x;
  const int lane = tid & 31, wave = tid >> 5;
  const int mlo = lane & 15, khalf = lane >> 4;

  floatx8 acc[2][4] = {};

  for (int k0 = 0; k0 < Kdim; k0 += BK) {
    // A tile: 128x32 bf16 via async-to-LDS b128 (4 segs / thread)
#pragma unroll
    for (int i = 0; i < 4; ++i) {
      int seg = tid + i * 128;
      int r = seg >> 2, c = (seg & 3) * 8;
      unsigned goff = (unsigned)(((m0 + r) * lda + k0 + c) * 2);
      async_load_b128(lds_off(&As[r][c]), goff, A);
    }
    // B tile: 32x64 bf16, coalesced global read, transposed LDS store
#pragma unroll
    for (int i = 0; i < 2; ++i) {
      int seg = tid + i * 128;
      int r = seg >> 3, c = (seg & 7) * 8;
      ushort8 v = *(const ushort8*)(Bw + (long)(k0 + r) * ldb + n0 + c);
#pragma unroll
      for (int e = 0; e < 8; ++e) BsT[c + e][r] = v[e];
    }
    wait_asynccnt0();
    __syncthreads();

    ABf a[2], b[4];
#pragma unroll
    for (int s = 0; s < 2; ++s) {
      const unsigned short* p = &As[wave * 32 + s * 16 + mlo][0];
      a[s].h[0] = *(const ushort8*)(p + khalf * 8);        // K 0..7 / 8..15
      a[s].h[1] = *(const ushort8*)(p + 16 + khalf * 8);   // K 16..23 / 24..31
    }
#pragma unroll
    for (int t = 0; t < 4; ++t) {
      const unsigned short* p = &BsT[t * 16 + mlo][0];
      b[t].h[0] = *(const ushort8*)(p + khalf * 8);
      b[t].h[1] = *(const ushort8*)(p + 16 + khalf * 8);
    }
#pragma unroll
    for (int s = 0; s < 2; ++s)
#pragma unroll
      for (int t = 0; t < 4; ++t)
        acc[s][t] = __builtin_amdgcn_wmma_f32_16x16x32_bf16(
            false, a[s].v, false, b[t].v, (short)0, acc[s][t], false, false);
    __syncthreads();
  }

#pragma unroll
  for (int s = 0; s < 2; ++s)
#pragma unroll
    for (int t = 0; t < 4; ++t)
#pragma unroll
      for (int r = 0; r < 8; ++r)
        C[(long)(m0 + wave * 32 + s * 16 + khalf * 8 + r) * ldc + n0 + t * 16 + mlo] =
            acc[s][t][r];
}

// ---------------------------------------------------------------------------
// Fused RMSNorm + RoPE + scale, fp32 in (qkv buffer [tok][16][256]) -> bf16 out.
// One block = one (token, head) row of 256; v path is plain convert.
// ---------------------------------------------------------------------------
__global__ __launch_bounds__(256) void normrope_kernel(
    const float* __restrict__ in, const float* __restrict__ scale,
    const int* __restrict__ positions, unsigned short* __restrict__ out,
    int headOffset, int nHeads, float outScale, int doNormRope) {
  const int row = blockIdx.x;
  const int tok = row / nHeads;
  const int hd  = row % nHeads;
  const float* src = in + ((long)tok * 16 + headOffset + hd) * cH;
  unsigned short* dst = out + ((long)tok * nHeads + hd) * cH;
  const int h = threadIdx.x;
  float v = src[h];

  if (!doNormRope) { dst[h] = f2bf(v); return; }

  __shared__ float buf[cH];
  __shared__ float red[8];
  float ss = v * v;
#pragma unroll
  for (int off = 16; off >= 1; off >>= 1) ss += __shfl_xor(ss, off, 32);
  if ((threadIdx.x & 31) == 0) red[threadIdx.x >> 5] = ss;
  __syncthreads();
  float tot = 0.f;
#pragma unroll
  for (int i = 0; i < 8; ++i) tot += red[i];
  const float inv = rsqrtf(tot * (1.0f / cH) + 1e-6f);
  v = v * inv * (1.0f + scale[h]);
  buf[h] = v;
  __syncthreads();

  if (h < cH / 2) {
    const float first = buf[h], second = buf[h + cH / 2];
    const float pos = (float)positions[tok];
    const float theta = pos * powf(10000.0f, -(float)h * (2.0f / cH));
    const float s = sinf(theta), c = cosf(theta);
    dst[h]          = f2bf((first * c - second * s) * outScale);
    dst[h + cH / 2] = f2bf((second * c + first * s) * outScale);
  }
}

// ---------------------------------------------------------------------------
// Flash attention, sliding-window causal (WINDOW=1024), GQA G=2.
// Grid: (T/64, N, B); 4 waves; wave owns 16 q rows.
// K chunk is DMA'd by the Tensor Data Mover (wave 0 issues one 64x512B tile
// with pad_enable reproducing the 264-ushort padded LDS rows) while all waves
// stage V transposed; Q A-fragments stay resident in VGPRs; P re-shaped
// C-layout -> A-layout through per-wave LDS.
// ---------------------------------------------------------------------------
__global__ __launch_bounds__(128) void attn_kernel(
    const unsigned short* __restrict__ qbf, const unsigned short* __restrict__ kbf,
    const unsigned short* __restrict__ vbf, unsigned short* __restrict__ encbf) {
  extern __shared__ char smemraw[];
  unsigned short* Ks = (unsigned short*)smemraw;   // [64][264]
  unsigned short* Vt = Ks + 64 * 264;              // [256][72]  (V transposed)
  unsigned short* Ps = Vt + 256 * 72;              // [4][16][64]

  const int qt = blockIdx.x, head = blockIdx.y, b = blockIdx.z;
  const int kvh = head >> 1;                       // G = 2
  const int tid = threadIdx.x, lane = tid & 31, wave = tid >> 5;
  const int mlo = lane & 15, khalf = lane >> 4;
  const int t0 = qt * 64;

  // resident Q fragments (8 ksteps over H=256)
  bf16x16 qa[8];
  {
    const int qrow = t0 + wave * 16 + mlo;
    const unsigned short* qp = qbf + (((long)b * cT + qrow) * cN + head) * cH;
#pragma unroll
    for (int kk = 0; kk < 8; ++kk) {
      ABf u;
      u.h[0] = *(const ushort8*)(qp + kk * 32 + khalf * 8);
      u.h[1] = *(const ushort8*)(qp + kk * 32 + 16 + khalf * 8);
      qa[kk] = u.v;
    }
  }

  float m_i[8], l_i[8];
#pragma unroll
  for (int r = 0; r < 8; ++r) { m_i[r] = -1e30f; l_i[r] = 0.f; }
  floatx8 Oacc[16] = {};

  int sstart = t0 - (cWIN - 1);
  if (sstart < 0) sstart = 0;
  sstart &= ~63;

  for (int s0 = sstart; s0 <= t0 + 63; s0 += 64) {
    const long chunk = (((long)b * cT + s0) * cK + kvh) * cH;

    // K chunk: one TDM descriptor moves 64 rows x 256 bf16 with global row
    // stride cK*cH, LDS-padded 4 DWORDs every 128 DWORDs (-> 264-short rows).
    if (wave == 0) {
      tdm_load_tile_2d(lds_off(Ks), kbf + chunk,
                       /*tensor_d0=*/cH, /*tensor_d1=*/64,
                       /*stride0=*/(unsigned long long)(cK * cH),
                       /*tile_d0=*/cH, /*tile_d1=*/64,
                       /*pad_interval(128 dw)=*/6u, /*pad_amount(4 dw)=*/3u, 1u);
    }

    // V chunk: transpose while staging (TDM cannot transpose)
#pragma unroll
    for (int i = 0; i < 16; ++i) {
      int seg = tid + i * 128;           // 0..2047
      int r = seg >> 5, c = (seg & 31) * 8;
      ushort8 vv = *(const ushort8*)(vbf + chunk + (long)r * cK * cH + c);
#pragma unroll
      for (int e = 0; e < 8; ++e) Vt[(c + e) * 72 + r] = vv[e];
    }
    if (wave == 0) __builtin_amdgcn_s_wait_tensorcnt(0);
    __syncthreads();

    // S = Q * K^T  (64 keys, contraction H=256)
    floatx8 sacc[4] = {};
#pragma unroll
    for (int kk = 0; kk < 8; ++kk)
#pragma unroll
      for (int nt = 0; nt < 4; ++nt) {
        ABf u;
        const unsigned short* p = &Ks[(nt * 16 + mlo) * 264 + kk * 32 + khalf * 8];
        u.h[0] = *(const ushort8*)(p);
        u.h[1] = *(const ushort8*)(p + 16);
        sacc[nt] = __builtin_amdgcn_wmma_f32_16x16x32_bf16(
            false, qa[kk], false, u.v, (short)0, sacc[nt], false, false);
      }

    // mask + online softmax (row lives in one 16-lane half per C layout)
    const int trow_base = t0 + wave * 16 + khalf * 8;
#pragma unroll
    for (int r = 0; r < 8; ++r) {
      const int t = trow_base + r;
      float mx = -3.0e38f;
#pragma unroll
      for (int nt = 0; nt < 4; ++nt) {
        const int s = s0 + nt * 16 + mlo;
        const bool ok = (s <= t) && ((t - s) < cWIN);
        const float val = ok ? sacc[nt][r] : cKMASK;
        sacc[nt][r] = val;
        mx = fmaxf(mx, val);
      }
#pragma unroll
      for (int off = 8; off >= 1; off >>= 1) mx = fmaxf(mx, __shfl_xor(mx, off, 16));
      const float newm = fmaxf(m_i[r], mx);
      const float alpha = expf(m_i[r] - newm);
      float rs = 0.f;
#pragma unroll
      for (int nt = 0; nt < 4; ++nt) {
        const float pv = expf(sacc[nt][r] - newm);
        sacc[nt][r] = pv;
        rs += pv;
      }
#pragma unroll
      for (int off = 8; off >= 1; off >>= 1) rs += __shfl_xor(rs, off, 16);
      l_i[r] = l_i[r] * alpha + rs;
      m_i[r] = newm;
#pragma unroll
      for (int ct = 0; ct < 16; ++ct) Oacc[ct][r] *= alpha;
    }

    // P: C-layout -> A-layout via per-wave LDS
#pragma unroll
    for (int r = 0; r < 8; ++r)
#pragma unroll
      for (int nt = 0; nt < 4; ++nt)
        Ps[wave * 1024 + (khalf * 8 + r) * 64 + nt * 16 + mlo] = f2bf(sacc[nt][r]);

    // O += P * V   (contraction 64 keys = 2 ksteps)
#pragma unroll
    for (int kk2 = 0; kk2 < 2; ++kk2) {
      ABf ua;
      const unsigned short* pp = &Ps[wave * 1024 + mlo * 64 + kk2 * 32];
      ua.h[0] = *(const ushort8*)(pp + khalf * 8);
      ua.h[1] = *(const ushort8*)(pp + 16 + khalf * 8);
#pragma unroll
      for (int ct = 0; ct < 16; ++ct) {
        ABf ub;
        const unsigned short* vp = &Vt[(ct * 16 + mlo) * 72 + kk2 * 32 + khalf * 8];
        ub.h[0] = *(const ushort8*)(vp);
        ub.h[1] = *(const ushort8*)(vp + 16);
        Oacc[ct] = __builtin_amdgcn_wmma_f32_16x16x32_bf16(
            false, ua.v, false, ub.v, (short)0, Oacc[ct], false, false);
      }
    }
    __syncthreads();
  }

  // epilogue: O / l, write bf16 enc[b][t][head][H]
#pragma unroll
  for (int ct = 0; ct < 16; ++ct)
#pragma unroll
    for (int r = 0; r < 8; ++r) {
      const int t = t0 + wave * 16 + khalf * 8 + r;
      const float val = Oacc[ct][r] / l_i[r];
      encbf[(((long)b * cT + t) * cN + head) * cH + ct * 16 + mlo] = f2bf(val);
    }
}

// ---------------------------------------------------------------------------
// Host orchestration.
// Workspace layout (bytes), total ~170 MB:
//   xbf 21M | wqkv 21M | wout 10.5M | qkvf32 67M | qbf 16.8M | kbf 8.4M |
//   vbf 8.4M | encbf 16.8M
// ---------------------------------------------------------------------------
extern "C" void kernel_launch(void* const* d_in, const int* in_sizes, int n_in,
                              void* d_out, int out_size, void* d_ws, size_t ws_size,
                              hipStream_t stream) {
  (void)in_sizes; (void)n_in; (void)out_size; (void)ws_size;
  const float* x      = (const float*)d_in[0];
  const float* q_w    = (const float*)d_in[1];
  const float* kv_w   = (const float*)d_in[2];
  const float* q_s    = (const float*)d_in[3];
  const float* k_s    = (const float*)d_in[4];
  const float* out_w  = (const float*)d_in[5];
  const int*   pos    = (const int*)d_in[6];
  // d_in[7] = attn_mask (recomputed arithmetically in-kernel)

  const long M = (long)cB * cT;                       // 4096
  char* w = (char*)d_ws;
  unsigned short* xbf    = (unsigned short*)(w);                              // 4096*2560
  unsigned short* wqkv   = (unsigned short*)(w += (long)M * cD * 2);          // 16*2560*256
  unsigned short* wout   = (unsigned short*)(w += 16L * cD * cH * 2);         // 2048*2560
  float*          qkvf   = (float*)(w += (long)cN * cH * cD * 2);             // 4096*16*256
  unsigned short* qbf    = (unsigned short*)(w += M * 16L * cH * 4);          // 4096*8*256
  unsigned short* kbf    = (unsigned short*)(w += M * (long)cN * cH * 2);     // 4096*4*256
  unsigned short* vbf    = (unsigned short*)(w += M * (long)cK * cH * 2);     // 4096*4*256
  unsigned short* encbf  = (unsigned short*)(w += M * (long)cK * cH * 2);     // 4096*2048

  // 1) bf16 conversions (weights fit in 192MB L2 once converted)
  f2bf_kernel<<<2048, 256, 0, stream>>>(x,     xbf,  M * cD);
  f2bf_kernel<<<2048, 256, 0, stream>>>(q_w,   wqkv, (long)cN * cD * cH);
  f2bf_kernel<<<2048, 256, 0, stream>>>(kv_w,  wqkv + (long)cN * cD * cH, 2L * cK * cD * cH);
  f2bf_kernel<<<2048, 256, 0, stream>>>(out_w, wout, (long)cN * cH * cD);

  // 2) QKV projection: one batched GEMM, z=0..7 q heads, 8..11 k, 12..15 v
  {
    dim3 grid(M / 128, cH / 64, 16);
    gemm_bf16_kernel<<<grid, 128, 0, stream>>>(
        xbf, 0L, cD,
        wqkv, (long)cD * cH, cH,
        qkvf, (long)cH, 16 * cH,
        cD);
  }

  // 3) RMSNorm + RoPE + 1/sqrt(H) (q), RMSNorm + RoPE (k), convert (v)
  normrope_kernel<<<(int)(M * cN), 256, 0, stream>>>(qkvf, q_s, pos, qbf, 0,  cN, 0.0625f, 1);
  normrope_kernel<<<(int)(M * cK), 256, 0, stream>>>(qkvf, k_s, pos, kbf, 8,  cK, 1.0f,    1);
  normrope_kernel<<<(int)(M * cK), 256, 0, stream>>>(qkvf, k_s, pos, vbf, 12, cK, 1.0f,    0);

  // 4) flash attention (dyn LDS: K 33792B + Vt 36864B + P 8192B = 78848B)
  {
    const int smem = (64 * 264 + 256 * 72 + 4 * 16 * 64) * 2;
    (void)hipFuncSetAttribute((const void*)attn_kernel,
                              hipFuncAttributeMaxDynamicSharedMemorySize, smem);
    dim3 grid(cT / 64, cN, cB);
    attn_kernel<<<grid, 128, smem, stream>>>(qbf, kbf, vbf, encbf);
  }

  // 5) output projection: [4096 x 2048] * [2048 x 2560] -> d_out fp32
  {
    dim3 grid(M / 128, cD / 64, 1);
    gemm_bf16_kernel<<<grid, 128, 0, stream>>>(
        encbf, 0L, cN * cH,
        wout, 0L, cD,
        (float*)d_out, 0L, cD,
        cN * cH);
  }
}
